// EmergencyGNNEnhanced_72112500900409
// MI455X (gfx1250) — matrix operand order repaired and checked by reading, not compiled, not run on hardware
//
#include <hip/hip_runtime.h>
#include <hip/hip_fp16.h>
#include <math.h>

typedef __attribute__((ext_vector_type(16))) _Float16 v16h;
typedef __attribute__((ext_vector_type(8)))  float    v8f;

enum { Nn = 100000, Ee = 1600000, El = 200000, D0 = 32, Dh = 256, Dh2 = 128, Dh4 = 64 };

// ---------------------------------------------------------------------------
// Small utility kernels
// ---------------------------------------------------------------------------
__global__ void fill_kernel(float* __restrict__ p, int n, float v) {
  int i = blockIdx.x * blockDim.x + threadIdx.x;
  if (i < n) p[i] = v;
}

__global__ void deg_count_kernel(const int* __restrict__ dst, float* __restrict__ deg, int E) {
  int i = blockIdx.x * blockDim.x + threadIdx.x;
  if (i < E) unsafeAtomicAdd(&deg[dst[i]], 1.0f);
}

__global__ void rsqrt_kernel(float* __restrict__ p, int n) {
  int i = blockIdx.x * blockDim.x + threadIdx.x;
  if (i < n) p[i] = rsqrtf(p[i]);   // deg >= 1 always (self-loops)
}

// Pack fp32 row-major W[K,Nc] into f16 WMMA B-fragment layout, kt-major tile
// order so the 4 consecutive n-tiles a wave needs at one kt are contiguous.
// tile = kt*ntiles + nt ; lane<16: K=j (j<8) / 8+j (j>=8), col=lane ; lane>=16: K=8+j (j<8) / 16+j (j>=8), col=lane-16
__global__ void pack_b_kernel(const float* __restrict__ W, _Float16* __restrict__ out, int K, int Nc) {
  int idx = blockIdx.x * blockDim.x + threadIdx.x;
  if (idx >= K * Nc) return;
  int j = idx & 15;
  int l = (idx >> 4) & 31;
  int tile = idx >> 9;
  int ntiles = Nc >> 4;
  int nt = tile % ntiles;
  int kt = tile / ntiles;
  int kk = (kt << 5) + ((j >> 3) << 4) + ((l >> 4) << 3) + (j & 7);
  int nn = (nt << 4) + (l & 15);
  out[idx] = (_Float16)W[(size_t)kk * Nc + nn];
}

// ---------------------------------------------------------------------------
// WMMA GEMM:  C[M,Nc] = act(A[M,K] @ W + bias)   (f16 inputs, f32 accumulate)
// One wave computes a 16x64 strip (4 C tiles) so each A fragment feeds 4 WMMAs.
// 8 waves (256 thr) per block = 128 rows x 64 cols per block; K stepped by 32.
// ---------------------------------------------------------------------------
template <bool BIAS, bool RELU>
__global__ __launch_bounds__(256) void gemm_wmma_kernel(
    const float* __restrict__ A, const _Float16* __restrict__ Bpk,
    const float* __restrict__ bias, float* __restrict__ C,
    int M, int K, int Nc)
{
  const int lane   = threadIdx.x & 31;
  const int tile_m = blockIdx.x * 8 + (threadIdx.x >> 5);
  if (tile_m * 16 >= M) return;                 // wave-uniform exit (EXEC all-1 inside)
  const int ntiles = Nc >> 4;
  const int nt0    = blockIdx.y << 2;           // 4 consecutive n-tiles per wave
  const int ktiles = K >> 5;

  // A fragment (16-bit A 16x32): lane<16 -> K base 0, lane>=16 -> +8;
  // elems 0..7 = K[kb..kb+7], elems 8..15 = K[kb+16..kb+23]
  const int   row    = tile_m * 16 + (lane & 15);
  const float* __restrict__ arow = A + (size_t)row * K;
  const int   kb_off = (lane >> 4) << 3;

  v8f c0 = {}, c1 = {}, c2 = {}, c3 = {};
  for (int kt = 0; kt < ktiles; ++kt) {
    const int kb = (kt << 5) + kb_off;
    float4 f0 = *(const float4*)(arow + kb);
    float4 f1 = *(const float4*)(arow + kb + 4);
    float4 f2 = *(const float4*)(arow + kb + 16);
    float4 f3 = *(const float4*)(arow + kb + 20);
    v16h a;
    a[0]  = (_Float16)f0.x; a[1]  = (_Float16)f0.y; a[2]  = (_Float16)f0.z; a[3]  = (_Float16)f0.w;
    a[4]  = (_Float16)f1.x; a[5]  = (_Float16)f1.y; a[6]  = (_Float16)f1.z; a[7]  = (_Float16)f1.w;
    a[8]  = (_Float16)f2.x; a[9]  = (_Float16)f2.y; a[10] = (_Float16)f2.z; a[11] = (_Float16)f2.w;
    a[12] = (_Float16)f3.x; a[13] = (_Float16)f3.y; a[14] = (_Float16)f3.z; a[15] = (_Float16)f3.w;

    const _Float16* __restrict__ bp =
        Bpk + ((size_t)(kt * ntiles + nt0) << 9) + lane * 16;
    v16h b0 = *(const v16h*)(bp);
    v16h b1 = *(const v16h*)(bp + 512);
    v16h b2 = *(const v16h*)(bp + 1024);
    v16h b3 = *(const v16h*)(bp + 1536);

    c0 = __builtin_amdgcn_wmma_f32_16x16x32_f16(false, a, false, b0, (short)0, c0, false, false);
    c1 = __builtin_amdgcn_wmma_f32_16x16x32_f16(false, a, false, b1, (short)0, c1, false, false);
    c2 = __builtin_amdgcn_wmma_f32_16x16x32_f16(false, a, false, b2, (short)0, c2, false, false);
    c3 = __builtin_amdgcn_wmma_f32_16x16x32_f16(false, a, false, b3, (short)0, c3, false, false);
  }

  // C/D layout: lane -> col = lane&15 ; VGPR r -> row = r + (lane>=16 ? 8 : 0)
  const int col0  = nt0 * 16 + (lane & 15);
  const int rbase = tile_m * 16 + ((lane >> 4) << 3);
  v8f acc[4] = {c0, c1, c2, c3};
  #pragma unroll
  for (int t = 0; t < 4; ++t) {
    const int col = col0 + t * 16;
    float bv = 0.0f;
    if (BIAS) bv = bias[col];
    #pragma unroll
    for (int r = 0; r < 8; ++r) {
      float v = acc[t][r] + bv;
      if (RELU) v = fmaxf(v, 0.0f);
      C[(size_t)(rbase + r) * Nc + col] = v;
    }
  }
}

static inline void launch_gemm(const float* A, const _Float16* Bpk, const float* bias,
                               float* C, int M, int K, int Nc, bool relu, hipStream_t s)
{
  dim3 grid((M / 16 + 7) / 8, Nc / 64);         // all layer Nc are multiples of 64
  if (bias) {
    if (relu) gemm_wmma_kernel<true, true ><<<grid, 256, 0, s>>>(A, Bpk, bias, C, M, K, Nc);
    else      gemm_wmma_kernel<true, false><<<grid, 256, 0, s>>>(A, Bpk, bias, C, M, K, Nc);
  } else {
    gemm_wmma_kernel<false, false><<<grid, 256, 0, s>>>(A, Bpk, nullptr, C, M, K, Nc);
  }
}

// ---------------------------------------------------------------------------
// GCN edge scatter: one wave per edge, lanes = 32 channels x 2
// ---------------------------------------------------------------------------
__global__ void gcn_scatter_kernel(const float* __restrict__ xw, const int* __restrict__ src,
                                   const int* __restrict__ dst, const float* __restrict__ dinv,
                                   float* __restrict__ agg, int E)
{
  int e    = (blockIdx.x * blockDim.x + threadIdx.x) >> 5;
  int lane = threadIdx.x & 31;
  if (e >= E) return;
  int   s = src[e], d = dst[e];
  float w = dinv[s] * dinv[d];
  float v0 = xw[(size_t)s * 64 + lane];
  float v1 = xw[(size_t)s * 64 + 32 + lane];
  unsafeAtomicAdd(&agg[(size_t)d * 64 + lane],      v0 * w);
  unsafeAtomicAdd(&agg[(size_t)d * 64 + 32 + lane], v1 * w);
}

// self-loop term + bias + relu (+ residual)
template <bool RESID>
__global__ void gcn_post_kernel(const float* __restrict__ agg, const float* __restrict__ xw,
                                const float* __restrict__ dinv, const float* __restrict__ bias,
                                float* __restrict__ h, int n64)
{
  int i = blockIdx.x * blockDim.x + threadIdx.x;
  if (i >= n64) return;
  int   node = i >> 6, c = i & 63;
  float di = dinv[node];
  float v  = agg[i] + xw[i] * di * di + bias[c];
  v = fmaxf(v, 0.0f);
  h[i] = RESID ? h[i] + v : v;
}

// ef[i] = concat(h64[sl[i]], h64[dl[i]])  (float4 granularity: 32 qwords per row)
__global__ void ef_gather_kernel(const float* __restrict__ h, const int* __restrict__ sl,
                                 const int* __restrict__ dl, float* __restrict__ ef, int EL)
{
  int idx = blockIdx.x * blockDim.x + threadIdx.x;
  if (idx >= EL * 32) return;
  int i = idx >> 5, q = idx & 31;
  const float4* h4 = (const float4*)h;
  float4 v = (q < 16) ? h4[(size_t)sl[i] * 16 + q] : h4[(size_t)dl[i] * 16 + (q - 16)];
  ((float4*)ef)[(size_t)i * 32 + q] = v;
}

// final head: 16 lanes per edge, coalesced 64B loads, shuffle-tree reduce
__global__ void predict_kernel(const float* __restrict__ e2, const float* __restrict__ w3,
                               const float* __restrict__ b3, float* __restrict__ out, int EL)
{
  int t = blockIdx.x * blockDim.x + threadIdx.x;
  int i = t >> 4;          // edge
  int j = t & 15;          // lane-in-group
  if (i >= EL) return;
  const float* row = e2 + (size_t)i * 64;
  float acc = 0.0f;
  #pragma unroll
  for (int it = 0; it < 4; ++it)
    acc += row[j + 16 * it] * w3[j + 16 * it];
  #pragma unroll
  for (int off = 8; off >= 1; off >>= 1)
    acc += __shfl_xor(acc, off, 32);            // stays within the 16-lane group
  if (j == 0) out[i] = 1.0f / (1.0f + __expf(-(acc + b3[0])));
}

// ---------------------------------------------------------------------------
extern "C" void kernel_launch(void* const* d_in, const int* in_sizes, int n_in,
                              void* d_out, int out_size, void* d_ws, size_t ws_size,
                              hipStream_t stream)
{
  const float* x       = (const float*)d_in[0];
  const int*   ei      = (const int*)  d_in[1];   // [2,E]: src row0, dst row1
  const int*   eli     = (const int*)  d_in[2];   // [2,EL]
  const float* enc_w1  = (const float*)d_in[3];
  const float* enc_b1  = (const float*)d_in[4];
  const float* enc_w2  = (const float*)d_in[5];
  const float* enc_b2  = (const float*)d_in[6];
  const float* conv_w0 = (const float*)d_in[7];
  const float* conv_b0 = (const float*)d_in[8];
  const float* conv_w1 = (const float*)d_in[9];
  const float* conv_b1 = (const float*)d_in[10];
  const float* conv_w2 = (const float*)d_in[11];
  const float* conv_b2 = (const float*)d_in[12];
  const float* ep_w1   = (const float*)d_in[13];
  const float* ep_b1   = (const float*)d_in[14];
  const float* ep_w2   = (const float*)d_in[15];
  const float* ep_b2   = (const float*)d_in[16];
  const float* ep_w3   = (const float*)d_in[17];
  const float* ep_b3   = (const float*)d_in[18];
  float* out = (float*)d_out;

  const int* e_src = ei;            const int* e_dst = ei + Ee;
  const int* l_src = eli;           const int* l_dst = eli + El;

  // ---- workspace carve (hipMalloc base is 256B aligned) ----
  char* ws = (char*)d_ws;
  size_t off = 0;
  auto alloc = [&](size_t bytes) -> void* {
    void* p = ws + off;
    off = (off + bytes + 255) & ~(size_t)255;
    return p;
  };
  float*    dinv  = (float*)alloc((size_t)Nn * 4);
  float*    h1    = (float*)alloc((size_t)Nn * Dh  * 4);   // reused as ef [El,128]
  float*    e1    = (float*)alloc((size_t)El * Dh2 * 4);
  float*    h128  = (float*)alloc((size_t)Nn * Dh2 * 4);   // reused as e2 [El,64]
  float*    xw    = (float*)alloc((size_t)Nn * Dh4 * 4);
  float*    agg   = (float*)alloc((size_t)Nn * Dh4 * 4);
  float*    h64   = (float*)alloc((size_t)Nn * Dh4 * 4);
  _Float16* w1pk  = (_Float16*)alloc((size_t)D0  * Dh  * 2);
  _Float16* w2pk  = (_Float16*)alloc((size_t)Dh  * Dh2 * 2);
  _Float16* c0pk  = (_Float16*)alloc((size_t)Dh2 * Dh4 * 2);
  _Float16* c1pk  = (_Float16*)alloc((size_t)Dh4 * Dh4 * 2);
  _Float16* c2pk  = (_Float16*)alloc((size_t)Dh4 * Dh4 * 2);
  _Float16* p1pk  = (_Float16*)alloc((size_t)Dh2 * Dh2 * 2);
  _Float16* p2pk  = (_Float16*)alloc((size_t)Dh2 * Dh4 * 2);
  float*    ef    = h1;
  float*    e2    = h128;

  auto cdiv = [](int a, int b) { return (a + b - 1) / b; };

  // ---- pack weights to f16 B-fragment layout ----
  pack_b_kernel<<<cdiv(D0  * Dh , 256), 256, 0, stream>>>(enc_w1,  w1pk, D0,  Dh);
  pack_b_kernel<<<cdiv(Dh  * Dh2, 256), 256, 0, stream>>>(enc_w2,  w2pk, Dh,  Dh2);
  pack_b_kernel<<<cdiv(Dh2 * Dh4, 256), 256, 0, stream>>>(conv_w0, c0pk, Dh2, Dh4);
  pack_b_kernel<<<cdiv(Dh4 * Dh4, 256), 256, 0, stream>>>(conv_w1, c1pk, Dh4, Dh4);
  pack_b_kernel<<<cdiv(Dh4 * Dh4, 256), 256, 0, stream>>>(conv_w2, c2pk, Dh4, Dh4);
  pack_b_kernel<<<cdiv(Dh2 * Dh2, 256), 256, 0, stream>>>(ep_w1,   p1pk, Dh2, Dh2);
  pack_b_kernel<<<cdiv(Dh2 * Dh4, 256), 256, 0, stream>>>(ep_w2,   p2pk, Dh2, Dh4);

  // ---- degree / dinv (self-loops => deg starts at 1) ----
  fill_kernel<<<cdiv(Nn, 256), 256, 0, stream>>>(dinv, Nn, 1.0f);
  deg_count_kernel<<<cdiv(Ee, 256), 256, 0, stream>>>(e_dst, dinv, Ee);
  rsqrt_kernel<<<cdiv(Nn, 256), 256, 0, stream>>>(dinv, Nn);

  // ---- node encoder ----
  launch_gemm(x,  w1pk, enc_b1, h1,   Nn, D0, Dh,  true, stream);   // [N,256]
  launch_gemm(h1, w2pk, enc_b2, h128, Nn, Dh, Dh2, true, stream);   // [N,128]

  // ---- GCN layer 0 (128 -> 64, no residual) ----
  launch_gemm(h128, c0pk, nullptr, xw, Nn, Dh2, Dh4, false, stream);
  fill_kernel<<<cdiv(Nn * Dh4, 256), 256, 0, stream>>>(agg, Nn * Dh4, 0.0f);
  gcn_scatter_kernel<<<cdiv(Ee * 32, 256), 256, 0, stream>>>(xw, e_src, e_dst, dinv, agg, Ee);
  gcn_post_kernel<false><<<cdiv(Nn * Dh4, 256), 256, 0, stream>>>(agg, xw, dinv, conv_b0, h64, Nn * Dh4);

  // ---- GCN layers 1,2 (64 -> 64, residual) ----
  launch_gemm(h64, c1pk, nullptr, xw, Nn, Dh4, Dh4, false, stream);
  fill_kernel<<<cdiv(Nn * Dh4, 256), 256, 0, stream>>>(agg, Nn * Dh4, 0.0f);
  gcn_scatter_kernel<<<cdiv(Ee * 32, 256), 256, 0, stream>>>(xw, e_src, e_dst, dinv, agg, Ee);
  gcn_post_kernel<true><<<cdiv(Nn * Dh4, 256), 256, 0, stream>>>(agg, xw, dinv, conv_b1, h64, Nn * Dh4);

  launch_gemm(h64, c2pk, nullptr, xw, Nn, Dh4, Dh4, false, stream);
  fill_kernel<<<cdiv(Nn * Dh4, 256), 256, 0, stream>>>(agg, Nn * Dh4, 0.0f);
  gcn_scatter_kernel<<<cdiv(Ee * 32, 256), 256, 0, stream>>>(xw, e_src, e_dst, dinv, agg, Ee);
  gcn_post_kernel<true><<<cdiv(Nn * Dh4, 256), 256, 0, stream>>>(agg, xw, dinv, conv_b2, h64, Nn * Dh4);

  // ---- edge predictor ----
  ef_gather_kernel<<<cdiv(El * 32, 256), 256, 0, stream>>>(h64, l_src, l_dst, ef, El);
  launch_gemm(ef, p1pk, ep_b1, e1, El, Dh2, Dh2, true, stream);     // [EL,128]
  launch_gemm(e1, p2pk, ep_b2, e2, El, Dh2, Dh4, true, stream);     // [EL,64]
  predict_kernel<<<cdiv(El * 16, 256), 256, 0, stream>>>(e2, ep_w3, ep_b3, out, El);
}
